// GCNLSTMRegressor_5403068859063
// MI455X (gfx1250) — matrix operand order repaired and checked by reading, not compiled, not run on hardware
//
#include <hip/hip_runtime.h>

#define BB   4
#define TT   192
#define LL   120
#define GG   500
#define HLH  500
#define NN   480      // B*L
#define H4   2000     // 4*HL
#define OUTC 24
#define HS   512      // padded hidden stride (zero-padded cols 500..511)

typedef __attribute__((ext_vector_type(16))) __bf16 v16bf;
typedef __attribute__((ext_vector_type(8)))  float  v8f;

__device__ __forceinline__ __bf16 f2bf(float f) { return (__bf16)f; }

__device__ __forceinline__ float sigm(float x) { return 1.0f / (1.0f + __expf(-x)); }

// Build a 16-bit A fragment from 4 float4 runs (ISA 16-bit A 16x32 layout):
// elements 0..7 <- K = base..base+7 ; elements 8..15 <- K = base+16..+23
__device__ __forceinline__ v16bf frag_from(float4 f0, float4 f1, float4 f2, float4 f3) {
  v16bf a;
  a[0]  = f2bf(f0.x); a[1]  = f2bf(f0.y); a[2]  = f2bf(f0.z); a[3]  = f2bf(f0.w);
  a[4]  = f2bf(f1.x); a[5]  = f2bf(f1.y); a[6]  = f2bf(f1.z); a[7]  = f2bf(f1.w);
  a[8]  = f2bf(f2.x); a[9]  = f2bf(f2.y); a[10] = f2bf(f2.z); a[11] = f2bf(f2.w);
  a[12] = f2bf(f3.x); a[13] = f2bf(f3.y); a[14] = f2bf(f3.z); a[15] = f2bf(f3.w);
  return a;
}

// A fragment for the H2 feature block: relu(pm*u[k] + qm*v[k])
__device__ __forceinline__ v16bf feat_frag(const float* __restrict__ u,
                                           const float* __restrict__ v,
                                           float pm, float qm, int kt, int half) {
  int kb = kt * 32 + 8 * half;
  float4 u0 = *(const float4*)(u + kb),      u1 = *(const float4*)(u + kb + 4);
  float4 u2 = *(const float4*)(u + kb + 16), u3 = *(const float4*)(u + kb + 20);
  float4 v0 = *(const float4*)(v + kb),      v1 = *(const float4*)(v + kb + 4);
  float4 v2 = *(const float4*)(v + kb + 16), v3 = *(const float4*)(v + kb + 20);
  float4 f0 = make_float4(fmaxf(pm*u0.x+qm*v0.x,0.f), fmaxf(pm*u0.y+qm*v0.y,0.f),
                          fmaxf(pm*u0.z+qm*v0.z,0.f), fmaxf(pm*u0.w+qm*v0.w,0.f));
  float4 f1 = make_float4(fmaxf(pm*u1.x+qm*v1.x,0.f), fmaxf(pm*u1.y+qm*v1.y,0.f),
                          fmaxf(pm*u1.z+qm*v1.z,0.f), fmaxf(pm*u1.w+qm*v1.w,0.f));
  float4 f2 = make_float4(fmaxf(pm*u2.x+qm*v2.x,0.f), fmaxf(pm*u2.y+qm*v2.y,0.f),
                          fmaxf(pm*u2.z+qm*v2.z,0.f), fmaxf(pm*u2.w+qm*v2.w,0.f));
  float4 f3 = make_float4(fmaxf(pm*u3.x+qm*v3.x,0.f), fmaxf(pm*u3.y+qm*v3.y,0.f),
                          fmaxf(pm*u3.z+qm*v3.z,0.f), fmaxf(pm*u3.w+qm*v3.w,0.f));
  return frag_from(f0, f1, f2, f3);
}

// A fragment straight from a zero-padded f32 row
__device__ __forceinline__ v16bf row_frag(const float* __restrict__ row, int kb) {
  float4 f0 = *(const float4*)(row + kb);
  float4 f1 = *(const float4*)(row + kb + 4);
  float4 f2 = *(const float4*)(row + kb + 16);
  float4 f3 = *(const float4*)(row + kb + 20);
  return frag_from(f0, f1, f2, f3);
}

// ---------------- GCN algebraic reduction ----------------
__global__ void uv_kernel(const float* __restrict__ W1, const float* __restrict__ W2,
                          float* __restrict__ u, float* __restrict__ v) {
  int g = blockIdx.x * blockDim.x + threadIdx.x;
  if (g >= 512) return;
  if (g >= GG) { u[g] = 0.f; v[g] = 0.f; return; }
  float su = 0.f, sv = 0.f;
  for (int gp = 0; gp < GG; ++gp) {
    float w  = W1[gp];
    float w2 = W2[gp * GG + g];
    su += fmaxf(w, 0.f) * w2;
    sv += fmaxf(-w, 0.f) * w2;
  }
  u[g] = su; v[g] = sv;
}

__global__ void pq_kernel(const float* __restrict__ x, const float* __restrict__ A,
                          float* __restrict__ p, float* __restrict__ q) {
  __shared__ float xr[LL];
  __shared__ float z1[LL];
  int bt  = blockIdx.x;
  int tid = threadIdx.x;
  if (tid < LL) xr[tid] = x[bt * LL + tid];
  __syncthreads();
  if (tid < LL) {
    float s = 0.f;
    for (int j = 0; j < LL; ++j) s += A[tid * LL + j] * xr[j];
    z1[tid] = s;
  }
  __syncthreads();
  if (tid < LL) {
    float sp = 0.f, sq = 0.f;
    for (int j = 0; j < LL; ++j) {
      float a = A[tid * LL + j];
      float z = z1[j];
      sp += a * fmaxf(z, 0.f);
      sq += a * fmaxf(-z, 0.f);
    }
    int b = bt / TT, t = bt % TT;
    p[t * NN + b * LL + tid] = sp;
    q[t * NN + b * LL + tid] = sq;
  }
}

// ---------------- weight packing into WMMA B-fragment layout ----------------
// idx = ((nt*Ktiles + kt)*32 + lane)*16 + i ; n = nt*16+(lane&15), k = kt*32+(i<8 ? (lane>>4)*8+i : 16+(lane>>4)*8+i-8)
__global__ void pack_comb_kernel(const float* __restrict__ Wih, const float* __restrict__ Whh,
                                 __bf16* __restrict__ out) {
  const int total = 125 * 32 * 512;
  for (int idx = blockIdx.x * blockDim.x + threadIdx.x; idx < total;
       idx += gridDim.x * blockDim.x) {
    int i    = idx & 15;
    int lane = (idx >> 4) & 31;
    int kt   = (idx >> 9) & 31;
    int nt   = idx >> 14;
    int k  = kt * 32 + ((i < 8) ? ((lane >> 4) * 8 + i) : (16 + (lane >> 4) * 8 + (i - 8)));
    int h4 = nt * 16 + (lane & 15);
    float val = 0.f;
    if (k < 512) { if (k < GG) val = Wih[h4 * GG + k]; }
    else         { int kk = k - 512; if (kk < HLH) val = Whh[h4 * HLH + kk]; }
    out[idx] = f2bf(val);
  }
}

__global__ void pack_b_kernel(const float* __restrict__ W, int K_src, int N_src,
                              int Ktiles, int Ntiles, __bf16* __restrict__ out) {
  int total = Ntiles * Ktiles * 512;
  for (int idx = blockIdx.x * blockDim.x + threadIdx.x; idx < total;
       idx += gridDim.x * blockDim.x) {
    int i    = idx & 15;
    int lane = (idx >> 4) & 31;
    int rem  = idx >> 9;
    int kt   = rem % Ktiles;
    int nt   = rem / Ktiles;
    int n = nt * 16 + (lane & 15);
    int k = kt * 32 + ((i < 8) ? ((lane >> 4) * 8 + i) : (16 + (lane >> 4) * 8 + (i - 8)));
    float val = (k < K_src && n < N_src) ? W[k * N_src + n] : 0.f;
    out[idx] = f2bf(val);
  }
}

// ---------------- LSTM gates: [H2_t | h_prev] @ [W_ih|W_hh]^T ----------------
// Ping-pong double buffering (no register copies -> no WMMA WAR hazard NOPs).
__global__ void lstm_gates_kernel(const float* __restrict__ p, const float* __restrict__ q,
                                  const float* __restrict__ u, const float* __restrict__ v,
                                  const float* __restrict__ hprev,
                                  const __bf16* __restrict__ Bp,
                                  const float* __restrict__ bih, const float* __restrict__ bhh,
                                  float* __restrict__ gates, int t) {
  const int KT = 32, NT = 5, Ngroups = 25;   // 125 N-tiles = 25 groups of 5
  int task = blockIdx.x * (blockDim.x >> 5) + (threadIdx.x >> 5);
  if (task >= 30 * Ngroups) return;
  int mt  = task / Ngroups;
  int nt0 = (task % Ngroups) * NT;
  int lane = threadIdx.x & 31;
  int half = lane >> 4;
  int lm   = lane & 15;
  int m    = mt * 16 + lm;
  float pm = p[t * NN + m];
  float qm = q[t * NN + m];
  const float* hr = hprev + m * HS;

  auto bload = [&](int s, int kt) {
    return *(const v16bf*)(Bp + ((((nt0 + s) * KT + kt) * 32 + lane) << 4));
  };

  v8f acc[NT] = {};
  v16bf b0[NT], b1[NT];
#pragma unroll
  for (int s = 0; s < NT; ++s) b0[s] = bload(s, 0);

  // feature phase: K tiles 0..15 (b0 holds fragment kt on loop entry)
  for (int kt = 0; kt < 16; kt += 2) {
#pragma unroll
    for (int s = 0; s < NT; ++s) b1[s] = bload(s, kt + 1);
    v16bf a = feat_frag(u, v, pm, qm, kt, half);
#pragma unroll
    for (int s = 0; s < NT; ++s)
      acc[s] = __builtin_amdgcn_wmma_f32_16x16x32_bf16(false, a, false, b0[s], (short)0,
                                                       acc[s], false, false);
#pragma unroll
    for (int s = 0; s < NT; ++s) b0[s] = bload(s, kt + 2);   // kt+2 <= 16 < 32, valid
    a = feat_frag(u, v, pm, qm, kt + 1, half);
#pragma unroll
    for (int s = 0; s < NT; ++s)
      acc[s] = __builtin_amdgcn_wmma_f32_16x16x32_bf16(false, a, false, b1[s], (short)0,
                                                       acc[s], false, false);
  }
  // hidden phase: K tiles 16..31, A from zero-padded h_prev rows
  for (int kt = 16; kt < 32; kt += 2) {
#pragma unroll
    for (int s = 0; s < NT; ++s) b1[s] = bload(s, kt + 1);
    v16bf a = row_frag(hr, (kt - 16) * 32 + 8 * half);
#pragma unroll
    for (int s = 0; s < NT; ++s)
      acc[s] = __builtin_amdgcn_wmma_f32_16x16x32_bf16(false, a, false, b0[s], (short)0,
                                                       acc[s], false, false);
    if (kt + 2 < 32) {
#pragma unroll
      for (int s = 0; s < NT; ++s) b0[s] = bload(s, kt + 2);
    }
    a = row_frag(hr, (kt - 15) * 32 + 8 * half);
#pragma unroll
    for (int s = 0; s < NT; ++s)
      acc[s] = __builtin_amdgcn_wmma_f32_16x16x32_bf16(false, a, false, b1[s], (short)0,
                                                       acc[s], false, false);
  }
#pragma unroll
  for (int s = 0; s < NT; ++s) {
#pragma unroll
    for (int r = 0; r < 8; ++r) {
      int mm = mt * 16 + r + 8 * half;
      int nn = (nt0 + s) * 16 + lm;
      gates[mm * H4 + nn] = acc[s][r] + bih[nn] + bhh[nn];
    }
  }
}

__global__ void lstm_update_kernel(const float* __restrict__ gates,
                                   float* __restrict__ c, float* __restrict__ h) {
  int idx = blockIdx.x * blockDim.x + threadIdx.x;
  if (idx >= NN * HLH) return;
  int n = idx / HLH, j = idx % HLH;
  const float* gr = gates + n * H4;
  float gi = gr[j], gf = gr[HLH + j], gg = gr[2 * HLH + j], go = gr[3 * HLH + j];
  int hc = n * HS + j;
  float cv = c[hc];
  cv = sigm(gf) * cv + sigm(gi) * tanhf(gg);
  c[hc] = cv;
  h[hc] = sigm(go) * tanhf(cv);
}

// ---------------- generic WMMA GEMM (MLP head), ping-pong A and B ----------------
// EPI 0: relu -> C[m*C_stride+n] ; EPI 2: sigmoid + scatter to (B,OUT,L)
// Ktiles must be even (16 / 94 / 32 / 94 here).
template <int EPI, int NT>
__global__ void wmma_gemm_kernel(const float* __restrict__ A, int A_stride,
                                 const __bf16* __restrict__ Bp, int Ktiles,
                                 const float* __restrict__ bias, int N_src,
                                 float* __restrict__ C, int C_stride,
                                 int Mtiles, int Ngroups) {
  int task = blockIdx.x * (blockDim.x >> 5) + (threadIdx.x >> 5);
  if (task >= Mtiles * Ngroups) return;
  int mt  = task / Ngroups;
  int nt0 = (task % Ngroups) * NT;
  int lane = threadIdx.x & 31;
  int half = lane >> 4;
  int lm   = lane & 15;
  const float* ar = A + (mt * 16 + lm) * A_stride + 8 * half;

  auto bload = [&](int s, int kt) {
    return *(const v16bf*)(Bp + ((((nt0 + s) * Ktiles + kt) * 32 + lane) << 4));
  };

  v8f acc[NT] = {};
  v16bf b0[NT], b1[NT];
#pragma unroll
  for (int s = 0; s < NT; ++s) b0[s] = bload(s, 0);
  v16bf a0 = row_frag(ar, 0);
  for (int kt = 0; kt < Ktiles; kt += 2) {
#pragma unroll
    for (int s = 0; s < NT; ++s) b1[s] = bload(s, kt + 1);
    v16bf a1 = row_frag(ar, (kt + 1) * 32);
#pragma unroll
    for (int s = 0; s < NT; ++s)
      acc[s] = __builtin_amdgcn_wmma_f32_16x16x32_bf16(false, a0, false, b0[s], (short)0,
                                                       acc[s], false, false);
    if (kt + 2 < Ktiles) {
#pragma unroll
      for (int s = 0; s < NT; ++s) b0[s] = bload(s, kt + 2);
      a0 = row_frag(ar, (kt + 2) * 32);
    }
#pragma unroll
    for (int s = 0; s < NT; ++s)
      acc[s] = __builtin_amdgcn_wmma_f32_16x16x32_bf16(false, a1, false, b1[s], (short)0,
                                                       acc[s], false, false);
  }
#pragma unroll
  for (int s = 0; s < NT; ++s) {
#pragma unroll
    for (int r = 0; r < 8; ++r) {
      int mm = mt * 16 + r + 8 * half;
      int nn = (nt0 + s) * 16 + lm;
      float val = acc[s][r] + ((nn < N_src) ? bias[nn] : 0.f);
      if (EPI == 0) {
        C[mm * C_stride + nn] = fmaxf(val, 0.f);
      } else if (EPI == 2) {
        if (nn < OUTC) {
          int b = mm / LL, l = mm % LL;
          C[(b * OUTC + nn) * LL + l] = sigm(val);
        }
      }
    }
  }
}

extern "C" void kernel_launch(void* const* d_in, const int* in_sizes, int n_in,
                              void* d_out, int out_size, void* d_ws, size_t ws_size,
                              hipStream_t stream) {
  const float* x    = (const float*)d_in[0];
  const float* Ahat = (const float*)d_in[1];
  const float* W1   = (const float*)d_in[2];
  const float* W2   = (const float*)d_in[3];
  const float* Wih  = (const float*)d_in[4];
  const float* Whh  = (const float*)d_in[5];
  const float* bih  = (const float*)d_in[6];
  const float* bhh  = (const float*)d_in[7];
  const float* Wh1  = (const float*)d_in[8];
  const float* bh1  = (const float*)d_in[9];
  const float* Wh2  = (const float*)d_in[10];
  const float* bh2  = (const float*)d_in[11];
  const float* Wh3  = (const float*)d_in[12];
  const float* bh3  = (const float*)d_in[13];
  const float* Wh4  = (const float*)d_in[14];
  const float* bh4  = (const float*)d_in[15];

  char* ws = (char*)d_ws;
  size_t off = 0;
  auto allocF = [&](size_t n) { float* r = (float*)(ws + off); off += n * sizeof(float); return r; };
  float* u     = allocF(512);
  float* v     = allocF(512);
  float* p     = allocF((size_t)TT * NN);
  float* q     = allocF((size_t)TT * NN);
  float* hb    = allocF((size_t)NN * HS);    // hb and cb contiguous -> one memset
  float* cb    = allocF((size_t)NN * HS);
  float* gates = allocF((size_t)NN * H4);
  float* za    = allocF((size_t)NN * 3008);
  float* zb    = allocF((size_t)NN * 1024);
  float* zc    = allocF((size_t)NN * 3008);
  off = (off + 31) & ~(size_t)31;
  auto allocB = [&](size_t n) { __bf16* r = (__bf16*)(ws + off); off += n * sizeof(__bf16); return r; };
  __bf16* WcombP = allocB((size_t)125 * 32 * 512);
  __bf16* Wh1P   = allocB((size_t)188 * 16 * 512);
  __bf16* Wh2P   = allocB((size_t)64  * 94 * 512);
  __bf16* Wh3P   = allocB((size_t)188 * 32 * 512);
  __bf16* Wh4P   = allocB((size_t)2   * 94 * 512);

  uv_kernel<<<2, 256, 0, stream>>>(W1, W2, u, v);
  pq_kernel<<<BB * TT, 128, 0, stream>>>(x, Ahat, p, q);
  {
    int elems = 125 * 32 * 512;
    pack_comb_kernel<<<(elems + 255) / 256, 256, 0, stream>>>(Wih, Whh, WcombP);
  }
  auto packLaunch = [&](const float* W, int Ks, int Ns, int Kt, int Nt, __bf16* out) {
    int elems = Nt * Kt * 512;
    pack_b_kernel<<<(elems + 255) / 256, 256, 0, stream>>>(W, Ks, Ns, Kt, Nt, out);
  };
  packLaunch(Wh1, 500, 3000, 16, 188, Wh1P);
  packLaunch(Wh2, 3000, 1000, 94, 64, Wh2P);   // 64 N-tiles -> zb fully written (pads = 0)
  packLaunch(Wh3, 1000, 3000, 32, 188, Wh3P);
  packLaunch(Wh4, 3000, 24, 94, 2, Wh4P);

  hipMemsetAsync(hb, 0, (size_t)2 * NN * HS * sizeof(float), stream);  // h0=c0=0, pads 0

  for (int t = 0; t < TT; ++t) {
    lstm_gates_kernel<<<(30 * 25 + 7) / 8, 256, 0, stream>>>(p, q, u, v, hb, WcombP,
                                                             bih, bhh, gates, t);
    lstm_update_kernel<<<(NN * HLH + 255) / 256, 256, 0, stream>>>(gates, cb, hb);
  }

  // MLP head: all A buffers stored with zero-padded strides -> no K guards
  {
    int tasks = 30 * 47;   // 188 N-tiles / NT=4
    wmma_gemm_kernel<0, 4><<<(tasks + 7) / 8, 256, 0, stream>>>(
        hb, HS, Wh1P, 16, bh1, 3000, za, 3008, 30, 47);
  }
  {
    int tasks = 30 * 16;   // 64 N-tiles / NT=4
    wmma_gemm_kernel<0, 4><<<(tasks + 7) / 8, 256, 0, stream>>>(
        za, 3008, Wh2P, 94, bh2, 1000, zb, 1024, 30, 16);
  }
  {
    int tasks = 30 * 47;
    wmma_gemm_kernel<0, 4><<<(tasks + 7) / 8, 256, 0, stream>>>(
        zb, 1024, Wh3P, 32, bh3, 3000, zc, 3008, 30, 47);
  }
  {
    int tasks = 30 * 1;    // 2 N-tiles / NT=2
    wmma_gemm_kernel<2, 2><<<(tasks + 7) / 8, 256, 0, stream>>>(
        zc, 3008, Wh4P, 94, bh4, 24, (float*)d_out, 0, 30, 1);
  }
  (void)in_sizes; (void)n_in; (void)out_size; (void)ws_size;
}
